// FAFormerEncoderLayer_88287347737057
// MI455X (gfx1250) — matrix operand order repaired and checked by reading, not compiled, not run on hardware
//
#include <hip/hip_runtime.h>
#include <hip/hip_bf16.h>
#include <math.h>

#define NN   4096
#define BB   4
#define NPERG 1024
#define KNB  16
#define DD   128
#define NKR  (NN*KNB)   // 65536 edge rows

typedef __attribute__((ext_vector_type(16))) __bf16 v16bf;
typedef __attribute__((ext_vector_type(8)))  float  v8f;

// ---------------------------------------------------------------- WMMA core
__device__ inline v8f wmma_bf16(v16bf a, v16bf b, v8f c) {
  return __builtin_amdgcn_wmma_f32_16x16x32_bf16(false, a, false, b, (short)0, c, false, false);
}

// A fragment (16x32 bf16) per documented CDNA5 layout:
// lanes 0-15: row M=lane; dwords 0..3 -> K=0..7, dwords 4..7 -> K=16..23
// lanes 16-31: row M=lane-16; dwords 0..3 -> K=8..15, dwords 4..7 -> K=24..31
// Per-lane element runs are contiguous in K -> compiler merges to ds_load_b128.
__device__ inline v16bf load_a_frag_bf(const __bf16* sA, int ld, int mBase, int kBase) {
  int lane = threadIdx.x & 31;
  int half = lane >> 4;
  int row  = mBase + (lane & 15);
  v16bf a;
#pragma unroll
  for (int e = 0; e < 16; ++e) {
    int d  = e >> 1;
    int kk = kBase + ((d < 4) ? 0 : 16) + half * 8 + 2 * (d & 3) + (e & 1);
    a[e] = sA[row * ld + kk];
  }
  return a;
}
__device__ inline v16bf load_a_frag_f32(const float* sA, int ld, int mBase, int kBase) {
  int lane = threadIdx.x & 31;
  int half = lane >> 4;
  int row  = mBase + (lane & 15);
  v16bf a;
#pragma unroll
  for (int e = 0; e < 16; ++e) {
    int d  = e >> 1;
    int kk = kBase + ((d < 4) ? 0 : 16) + half * 8 + 2 * (d & 3) + (e & 1);
    a[e] = (__bf16)sA[row * ld + kk];
  }
  return a;
}
// B fragment (32x16 bf16): lane holds column N = nBase + lane%16, K = (lane/16)*8 + 0..7 and 16 + (lane/16)*8 + 0..7.
// B tile is stored TRANSPOSED in LDS: sBt[col*32 + kk] -> two 16B runs per lane.
__device__ inline v16bf load_b_frag(const __bf16* sBt, int nBase) {
  int lane = threadIdx.x & 31;
  const __bf16* p = sBt + (nBase + (lane & 15)) * 32 + (lane >> 4) * 8;
  v16bf b;
#pragma unroll
  for (int e = 0; e < 16; ++e) b[e] = p[(e < 8) ? e : (e + 8)];
  return b;
}

// ---------------------------------------------------------------- generic GEMM
// C(MxNd) = A(MxKd) @ B(Kd x Nd) + bias + optional Cadd.
// REQUIRES: M % 128 == 0, Nd % 128 == 0, Kd % 32 == 0 (all launches satisfy this).
// asrc: 0 plain A; 1 em-pair gather [tok(n) | tok(nidx) | ff(nk)]; 2 cat2 [aux0 | aux1]
__global__ __launch_bounds__(256) void gemm_kernel(
    const float* __restrict__ A, long lda,
    const float* __restrict__ Bw, int ldb,
    const float* __restrict__ bias, const float* __restrict__ Cadd,
    float* __restrict__ C, int ldc,
    int M, int Nd, int Kd,
    int asrc, const float* __restrict__ aux0, const float* __restrict__ aux1,
    const int* __restrict__ nidx)
{
  __shared__ __bf16 sA[128 * 32];
  __shared__ __bf16 sBt[128 * 32];   // transposed: [col][kk]
  int tid = threadIdx.x;
  int bm = blockIdx.x, bn = blockIdx.y;
  int wave = tid >> 5, wm = wave >> 2, wn = wave & 3;
  int lane = tid & 31, half = lane >> 4;

  v8f acc[4][2];
#pragma unroll
  for (int sm = 0; sm < 4; ++sm)
#pragma unroll
    for (int sn = 0; sn < 2; ++sn)
#pragma unroll
      for (int e = 0; e < 8; ++e) acc[sm][sn][e] = 0.f;

  // A-stage geometry: thread covers 16 consecutive K elements of one row-half.
  int arow = tid >> 1;                 // 0..127
  int akk0 = (tid & 1) * 16;           // 0 or 16
  int agr  = bm * 128 + arow;

  for (int k0 = 0; k0 < Kd; k0 += 32) {
    // ---- stage A tile 128x32 (uniform source selection per k0) ----
    if (asrc == 0) {
      const float* src = A + (long)agr * lda + k0 + akk0;
#pragma unroll
      for (int i = 0; i < 16; ++i) sA[arow * 32 + akk0 + i] = (__bf16)src[i];
    } else if (asrc == 1) {
      int n = agr >> 4, kb = agr & 15;
      const float* src;
      if (k0 < DD)            src = aux0 + (long)n * DD + k0 + akk0;
      else if (k0 < 2 * DD)   src = aux0 + (long)nidx[n * KNB + kb] * DD + (k0 - DD) + akk0;
      else                    src = aux1 + (long)agr * DD + (k0 - 2 * DD) + akk0;
#pragma unroll
      for (int i = 0; i < 16; ++i) sA[arow * 32 + akk0 + i] = (__bf16)src[i];
    } else {
      const float* src = (k0 < DD) ? (aux0 + (long)agr * DD + k0 + akk0)
                                   : (aux1 + (long)agr * DD + (k0 - DD) + akk0);
#pragma unroll
      for (int i = 0; i < 16; ++i) sA[arow * 32 + akk0 + i] = (__bf16)src[i];
    }
    // ---- stage B tile 32x128, transposed into LDS ----
#pragma unroll
    for (int i = 0; i < 16; ++i) {
      int idx = tid * 16 + i;
      int kk = idx >> 7, col = idx & 127;          // coalesced global read
      sBt[col * 32 + kk] = (__bf16)Bw[(long)(k0 + kk) * ldb + bn * 128 + col];
    }
    if (k0 + 32 < Kd)
      __builtin_prefetch(Bw + (long)(k0 + 32) * ldb + bn * 128 + (tid & 127), 0, 1);
    __syncthreads();
    v16bf aF[4], bF[2];
#pragma unroll
    for (int sm = 0; sm < 4; ++sm) aF[sm] = load_a_frag_bf(sA, 32, wm * 64 + sm * 16, 0);
#pragma unroll
    for (int sn = 0; sn < 2; ++sn) bF[sn] = load_b_frag(sBt, wn * 32 + sn * 16);
#pragma unroll
    for (int sm = 0; sm < 4; ++sm)
#pragma unroll
      for (int sn = 0; sn < 2; ++sn) acc[sm][sn] = wmma_bf16(aF[sm], bF[sn], acc[sm][sn]);
    __syncthreads();
  }
#pragma unroll
  for (int sm = 0; sm < 4; ++sm)
#pragma unroll
    for (int sn = 0; sn < 2; ++sn) {
      int col = bn * 128 + wn * 32 + sn * 16 + (lane & 15);
      float badd = bias ? bias[col] : 0.f;
#pragma unroll
      for (int r = 0; r < 8; ++r) {
        int row = bm * 128 + wm * 64 + sm * 16 + half * 8 + r;
        float v = acc[sm][sn][r] + badd;
        if (Cadd) v += Cadd[(long)row * ldc + col];
        C[(long)row * ldc + col] = v;
      }
    }
}

// ---------------------------------------------------------------- row GELU+LN
__global__ __launch_bounds__(128) void rowln_kernel(
    const float* __restrict__ src, const float* __restrict__ g, const float* __restrict__ b,
    float* __restrict__ dst, int W, int dogelu)
{
  __shared__ float red[128];
  long row = blockIdx.x;
  int tid = threadIdx.x;
  const float* x = src + row * (long)W;
  float* y = dst + row * (long)W;
  int per = W >> 7;                         // 1 (W=128) or 4 (W=512)
  float v[4];
  float s = 0.f;
  for (int i = 0; i < per; ++i) {
    float t = x[tid + i * 128];
    if (dogelu) t = 0.5f * t * (1.f + erff(t * 0.70710678f));
    v[i] = t; s += t;
  }
  red[tid] = s; __syncthreads();
  for (int off = 64; off; off >>= 1) { if (tid < off) red[tid] += red[tid + off]; __syncthreads(); }
  float mn = red[0] / (float)W; __syncthreads();
  float q = 0.f;
  for (int i = 0; i < per; ++i) { float d = v[i] - mn; q += d * d; }
  red[tid] = q; __syncthreads();
  for (int off = 64; off; off >>= 1) { if (tid < off) red[tid] += red[tid + off]; __syncthreads(); }
  float rs = rsqrtf(red[0] / (float)W + 1e-5f);
  for (int i = 0; i < per; ++i) {
    int c = tid + i * 128;
    y[c] = (v[i] - mn) * rs * g[c] + b[c];
  }
}

// ---------------------------------------------------------------- attention
__global__ __launch_bounds__(128) void attn_softmax_kernel(
    const float* __restrict__ qkv, const float* __restrict__ qkve,
    const int* __restrict__ nidx, const unsigned char* __restrict__ nmask,
    const float* __restrict__ mw, const float* __restrict__ ew,
    float* __restrict__ attnW)
{
  int n = blockIdx.x;
  int t = threadIdx.x;
  int h = t >> 4, k = t & 15;
  const float* q  = qkv  + (long)n * 384 + h * 16;
  int nb = nidx[n * KNB + k];
  const float* kv = qkv  + (long)nb * 384 + 128 + h * 16;
  const float* qe = qkve + ((long)n * KNB + k) * 256 + h * 16;
  float s = 0.f;
#pragma unroll
  for (int d = 0; d < 16; ++d) s += (q[d] + kv[d]) * mw[d] + qe[d] * ew[d];
  if (!nmask[n * KNB + k]) s = -1e9f;
  float m = s;
  for (int off = 8; off; off >>= 1) m = fmaxf(m, __shfl_xor(m, off));
  float e = expf(s - m);
  float sum = e;
  for (int off = 8; off; off >>= 1) sum += __shfl_xor(sum, off);
  attnW[(long)n * 128 + h * 16 + k] = e / sum;
}

__global__ __launch_bounds__(256) void context_kernel(
    const float* __restrict__ qkv, const float* __restrict__ qkve,
    const int* __restrict__ nidx, const float* __restrict__ attnW,
    float* __restrict__ scec)
{
  int n = blockIdx.x, t = threadIdx.x;
  int hd = t & 127, h = hd >> 4, d = hd & 15;
  const float* aw = attnW + (long)n * 128 + h * 16;
  float s = 0.f;
  if (t < 128) {
#pragma unroll
    for (int k = 0; k < KNB; ++k)
      s += aw[k] * qkv[(long)nidx[n * KNB + k] * 384 + 256 + h * 16 + d];
  } else {
#pragma unroll
    for (int k = 0; k < KNB; ++k)
      s += aw[k] * qkve[((long)n * KNB + k) * 256 + 128 + h * 16 + d];
  }
  scec[(long)n * 256 + t] = s;
}

__global__ __launch_bounds__(256) void gate_kernel(
    const float* __restrict__ tok, const float* __restrict__ gw, const float* __restrict__ gb,
    float* __restrict__ gate)
{
  int n = blockIdx.x * 8 + (threadIdx.x >> 5);
  int lane = threadIdx.x & 31;
  float s = 0.f;
#pragma unroll
  for (int i = 0; i < 4; ++i) { int c = lane * 4 + i; s += tok[(long)n * DD + c] * gw[c]; }
  for (int off = 16; off; off >>= 1) s += __shfl_xor(s, off);
  if (lane == 0) gate[n] = 1.f / (1.f + expf(-(s + gb[0])));
}

// ---------------------------------------------------------------- frames / eigh
__device__ void jacobi3(const float cv[6], float Vout[9]) {
  float A[3][3] = {{cv[0], cv[1], cv[2]}, {cv[1], cv[3], cv[4]}, {cv[2], cv[4], cv[5]}};
  float Q[3][3] = {{1, 0, 0}, {0, 1, 0}, {0, 0, 1}};
  for (int sweep = 0; sweep < 10; ++sweep) {
    for (int pi = 0; pi < 3; ++pi) {
      int p = (pi == 2) ? 1 : 0;
      int q = (pi == 0) ? 1 : 2;
      float apq = A[p][q];
      if (fabsf(apq) < 1e-20f) continue;
      float th = 0.5f * (A[q][q] - A[p][p]) / apq;
      float t = ((th >= 0.f) ? 1.f : -1.f) / (fabsf(th) + sqrtf(th * th + 1.f));
      float c = rsqrtf(t * t + 1.f);
      float s = t * c;
      for (int r = 0; r < 3; ++r) { float a = A[r][p], b = A[r][q]; A[r][p] = c * a - s * b; A[r][q] = s * a + c * b; }
      for (int r = 0; r < 3; ++r) { float a = A[p][r], b = A[q][r]; A[p][r] = c * a - s * b; A[q][r] = s * a + c * b; }
      for (int r = 0; r < 3; ++r) { float a = Q[r][p], b = Q[r][q]; Q[r][p] = c * a - s * b; Q[r][q] = s * a + c * b; }
    }
  }
  float ev[3] = {A[0][0], A[1][1], A[2][2]};
  int o0 = 0, o1 = 1, o2 = 2, tmp;
  if (ev[o0] > ev[o1]) { tmp = o0; o0 = o1; o1 = tmp; }
  if (ev[o1] > ev[o2]) { tmp = o1; o1 = o2; o2 = tmp; }
  if (ev[o0] > ev[o1]) { tmp = o0; o0 = o1; o1 = tmp; }
  int ord[3] = {o0, o1, o2};
  for (int i = 0; i < 3; ++i)
    for (int j = 0; j < 3; ++j) Vout[i * 3 + j] = Q[j][ord[i]];   // row i = eigvec i (ascending)
}

__global__ __launch_bounds__(256) void center_kernel(const float* __restrict__ geo, float* __restrict__ cOut) {
  __shared__ float red[256];
  int b = blockIdx.x, tid = threadIdx.x;
  for (int comp = 0; comp < 3; ++comp) {
    float s = 0.f;
    for (int p = tid; p < NPERG; p += 256) s += geo[(long)(b * NPERG + p) * 3 + comp];
    red[tid] = s; __syncthreads();
    for (int off = 128; off; off >>= 1) { if (tid < off) red[tid] += red[tid + off]; __syncthreads(); }
    if (tid == 0) cOut[b * 3 + comp] = red[0] / (float)NPERG;
    __syncthreads();
  }
}

__global__ __launch_bounds__(256) void graph_frames_kernel(
    const float* __restrict__ geo, float* __restrict__ Vg, float* __restrict__ c2)
{
  __shared__ float red[256];
  __shared__ float cen[3];
  __shared__ float cv[6];
  int b = blockIdx.x, tid = threadIdx.x;
  for (int comp = 0; comp < 3; ++comp) {
    float s = 0.f;
    for (int p = tid; p < NPERG; p += 256) s += geo[(long)(b * NPERG + p) * 3 + comp];
    red[tid] = s; __syncthreads();
    for (int off = 128; off; off >>= 1) { if (tid < off) red[tid] += red[tid + off]; __syncthreads(); }
    if (tid == 0) cen[comp] = red[0] / (float)NPERG;
    __syncthreads();
  }
  const int ci[6] = {0, 0, 0, 1, 1, 2};
  const int cj[6] = {0, 1, 2, 1, 2, 2};
  for (int c = 0; c < 6; ++c) {
    float s = 0.f;
    for (int p = tid; p < NPERG; p += 256) {
      const float* x = geo + (long)(b * NPERG + p) * 3;
      s += (x[ci[c]] - cen[ci[c]]) * (x[cj[c]] - cen[cj[c]]);
    }
    red[tid] = s; __syncthreads();
    for (int off = 128; off; off >>= 1) { if (tid < off) red[tid] += red[tid + off]; __syncthreads(); }
    if (tid == 0) cv[c] = red[0];
    __syncthreads();
  }
  if (tid == 0) {
    float V[9];
    jacobi3(cv, V);
    for (int i = 0; i < 9; ++i) Vg[b * 9 + i] = V[i];
    for (int i = 0; i < 3; ++i) c2[b * 3 + i] = cen[i];
  }
}

__global__ __launch_bounds__(256) void geo_out_kernel(
    const float* __restrict__ geo, const float* __restrict__ gate,
    const float* __restrict__ c0, float* __restrict__ outGeo)
{
  int n = blockIdx.x * blockDim.x + threadIdx.x;
  if (n >= NN) return;
  int b = n >> 10;
  float g = gate[n];
  for (int i = 0; i < 3; ++i)
    outGeo[n * 3 + i] = c0[b * 3 + i] * g + geo[n * 3 + i] * (1.f - g);
}

__global__ __launch_bounds__(256) void projg_kernel(
    const float* __restrict__ geo, const float* __restrict__ Vg, const float* __restrict__ c2,
    float* __restrict__ projg)
{
  int n = blockIdx.x * blockDim.x + threadIdx.x;
  if (n >= NN) return;
  int b = n >> 10;
  float dx = geo[n * 3 + 0] - c2[b * 3 + 0];
  float dy = geo[n * 3 + 1] - c2[b * 3 + 1];
  float dz = geo[n * 3 + 2] - c2[b * 3 + 2];
  const float* V = Vg + b * 9;
  for (int i = 0; i < 3; ++i)
    projg[n * 3 + i] = V[i * 3 + 0] * dx + V[i * 3 + 1] * dy + V[i * 3 + 2] * dz;
}

__global__ __launch_bounds__(256) void node_frames_kernel(
    const float* __restrict__ geo, const int* __restrict__ nidx,
    const unsigned char* __restrict__ nmask,
    float* __restrict__ proj, float* __restrict__ rnormb)
{
  int n = blockIdx.x * blockDim.x + threadIdx.x;
  if (n >= NN) return;
  float gx = geo[n * 3], gy = geo[n * 3 + 1], gz = geo[n * 3 + 2];
  float cx = 0.f, cy = 0.f, cz = 0.f, cnt = 0.f;
  for (int k = 0; k < KNB; ++k) {
    int nb = nidx[n * KNB + k];
    float rx = gx - geo[nb * 3], ry = gy - geo[nb * 3 + 1], rz = gz - geo[nb * 3 + 2];
    float mm = nmask[n * KNB + k] ? 1.f : 0.f;
    rnormb[n * KNB + k] = rx * rx + ry * ry + rz * rz;
    cx += rx * mm; cy += ry * mm; cz += rz * mm; cnt += mm;
  }
  float inv = (cnt > 0.f) ? 1.f / cnt : 0.f;
  cx *= inv; cy *= inv; cz *= inv;
  float s00 = 0, s01 = 0, s02 = 0, s11 = 0, s12 = 0, s22 = 0;
  for (int k = 0; k < KNB; ++k) {
    int nb = nidx[n * KNB + k];
    float mm = nmask[n * KNB + k] ? 1.f : 0.f;
    float dx = (gx - geo[nb * 3]     - cx) * mm;
    float dy = (gy - geo[nb * 3 + 1] - cy) * mm;
    float dz = (gz - geo[nb * 3 + 2] - cz) * mm;
    s00 += dx * dx; s01 += dx * dy; s02 += dx * dz;
    s11 += dy * dy; s12 += dy * dz; s22 += dz * dz;
  }
  float cv[6] = {s00, s01, s02, s11, s12, s22};
  float V[9];
  jacobi3(cv, V);
  for (int k = 0; k < KNB; ++k) {
    int nb = nidx[n * KNB + k];
    float mm = nmask[n * KNB + k] ? 1.f : 0.f;
    float dx = (gx - geo[nb * 3]     - cx) * mm;
    float dy = (gy - geo[nb * 3 + 1] - cy) * mm;
    float dz = (gz - geo[nb * 3 + 2] - cz) * mm;
    long p = ((long)n * KNB + k) * 3;
    proj[p + 0] = V[0] * dx + V[1] * dy + V[2] * dz;
    proj[p + 1] = V[3] * dx + V[4] * dy + V[5] * dz;
    proj[p + 2] = V[6] * dx + V[7] * dy + V[8] * dz;
  }
}

// ---------------------------------------------------------------- fused frame MLP
// Rows ordered (group, frame): 8 groups x 8 frames = 64 rows/block.
// in -> GEMM1(w1) -> +b1 -> GELU -> LN(g,be) -> GEMM2(w2) -> mean over 8 frames -> +b2
// mode 0: coord (feat = [OPS*proj, rnorm], group=(n,k)); mode 1: wf (feat = OPS*projg, group=n)
__global__ __launch_bounds__(256) void frame_mlp_kernel(
    int mode, const float* __restrict__ proj, const float* __restrict__ rnormb,
    const float* __restrict__ w1, const float* __restrict__ b1,
    const float* __restrict__ g, const float* __restrict__ be,
    const float* __restrict__ w2, const float* __restrict__ b2,
    float* __restrict__ out)
{
  __shared__ __bf16 sA[64 * 32];
  __shared__ __bf16 sBt[128 * 32];   // transposed: [col][kk]
  __shared__ float  sH[64 * 128];
  __shared__ float  sF[64 * 4];
  int tid = threadIdx.x;
  int g0 = blockIdx.x * 8;

  if (tid < 64) {
    int r = tid, grp = g0 + (r >> 3), f = r & 7;
    float s0 = ((f >> 2) & 1) ? 1.f : -1.f;
    float s1 = ((f >> 1) & 1) ? 1.f : -1.f;
    float s2 = (f & 1) ? 1.f : -1.f;
    sF[r * 4 + 0] = s0 * proj[(long)grp * 3 + 0];
    sF[r * 4 + 1] = s1 * proj[(long)grp * 3 + 1];
    sF[r * 4 + 2] = s2 * proj[(long)grp * 3 + 2];
    sF[r * 4 + 3] = (mode == 0) ? rnormb[grp] : 0.f;
  }
  int F = (mode == 0) ? 4 : 3;
#pragma unroll
  for (int i = 0; i < 16; ++i) {               // stage w1 (Fx128 zero-padded) transposed
    int idx = tid * 16 + i;
    int kk = idx >> 7, col = idx & 127;
    sBt[col * 32 + kk] = (__bf16)((kk < F) ? w1[kk * 128 + col] : 0.f);
  }
  __syncthreads();
#pragma unroll
  for (int i = 0; i < 8; ++i) {                // stage A (64x4 zero-padded to 64x32)
    int idx = tid * 8 + i;
    int row = idx >> 5, kk = idx & 31;
    sA[row * 32 + kk] = (__bf16)((kk < 4) ? sF[row * 4 + kk] : 0.f);
  }
  __syncthreads();

  int wave = tid >> 5, m = wave >> 1, ch = wave & 1;
  int lane = tid & 31, half = lane >> 4;
  {
    v16bf aF = load_a_frag_bf(sA, 32, m * 16, 0);
#pragma unroll
    for (int sn = 0; sn < 4; ++sn) {
      v16bf bF = load_b_frag(sBt, (ch * 4 + sn) * 16);
      v8f z;
#pragma unroll
      for (int e = 0; e < 8; ++e) z[e] = 0.f;
      v8f r = wmma_bf16(aF, bF, z);
      int col = (ch * 4 + sn) * 16 + (lane & 15);
#pragma unroll
      for (int rr = 0; rr < 8; ++rr) {
        int row = m * 16 + half * 8 + rr;
        sH[row * 128 + col] = r[rr] + b1[col];
      }
    }
  }
  __syncthreads();
  if (tid < 64) {                              // GELU + LN over 128-wide rows
    int row = tid;
    float s = 0.f;
    for (int c = 0; c < 128; ++c) {
      float t = sH[row * 128 + c];
      t = 0.5f * t * (1.f + erff(t * 0.70710678f));
      sH[row * 128 + c] = t; s += t;
    }
    float mn = s / 128.f, vv = 0.f;
    for (int c = 0; c < 128; ++c) { float d = sH[row * 128 + c] - mn; vv += d * d; }
    float rs = rsqrtf(vv / 128.f + 1e-5f);
    for (int c = 0; c < 128; ++c)
      sH[row * 128 + c] = (sH[row * 128 + c] - mn) * rs * g[c] + be[c];
  }
  __syncthreads();

  v8f acc2[4];
#pragma unroll
  for (int sn = 0; sn < 4; ++sn)
#pragma unroll
    for (int e = 0; e < 8; ++e) acc2[sn][e] = 0.f;

  for (int k0 = 0; k0 < 128; k0 += 32) {
    __syncthreads();
#pragma unroll
    for (int i = 0; i < 16; ++i) {             // stage w2 tile 32x128 transposed
      int idx = tid * 16 + i;
      int kk = idx >> 7, col = idx & 127;
      sBt[col * 32 + kk] = (__bf16)w2[(k0 + kk) * 128 + col];
    }
    __syncthreads();
    v16bf aF = load_a_frag_f32(sH, 128, m * 16, k0);
#pragma unroll
    for (int sn = 0; sn < 4; ++sn)
      acc2[sn] = wmma_bf16(aF, load_b_frag(sBt, (ch * 4 + sn) * 16), acc2[sn]);
  }
#pragma unroll
  for (int sn = 0; sn < 4; ++sn) {             // mean over 8 consecutive frame-rows
    int col = (ch * 4 + sn) * 16 + (lane & 15);
    float s = 0.f;
#pragma unroll
    for (int rr = 0; rr < 8; ++rr) s += acc2[sn][rr];
    out[(long)(g0 + m * 2 + half) * 128 + col] = s * 0.125f + b2[col];
  }
}

// ---------------------------------------------------------------- edge gating
__global__ __launch_bounds__(128) void edge_gate_kernel(
    const float* __restrict__ em_out, const float* __restrict__ edge,
    const float* __restrict__ ew, const float* __restrict__ eb,
    float* __restrict__ outEdge)
{
  __shared__ float red[128];
  __shared__ float sg;
  long row = blockIdx.x;
  int tid = threadIdx.x;
  float v = em_out[row * 128 + tid];
  red[tid] = v * ew[tid]; __syncthreads();
  for (int off = 64; off; off >>= 1) { if (tid < off) red[tid] += red[tid + off]; __syncthreads(); }
  if (tid == 0) sg = 1.f / (1.f + expf(-(red[0] + eb[0])));
  __syncthreads();
  outEdge[row * 128 + tid] = edge[row * 128 + tid] + v * sg;
}

// ---------------------------------------------------------------- host launcher
extern "C" void kernel_launch(void* const* d_in, const int* in_sizes, int n_in,
                              void* d_out, int out_size, void* d_ws, size_t ws_size,
                              hipStream_t stream) {
  (void)in_sizes; (void)n_in; (void)out_size; (void)ws_size;
  const float* tok  = (const float*)d_in[0];
  const float* geo  = (const float*)d_in[1];
  const float* edge = (const float*)d_in[2];
  const int*   nidx = (const int*)d_in[3];
  const unsigned char* nmask = (const unsigned char*)d_in[5];
#define PR(i) ((const float*)d_in[6 + (i)])
  // 0 attn_ln_g 1 attn_ln_b 2 qkv_w 3 qkv_b 4 lne_g 5 lne_b 6 qkve_w 7 qkve_b
  // 8 mlp_attn_w 9 edge_attn_w 10..15 out MLP 16 gate_w 17 gate_b 18 fagg_w 19 fagg_b
  // 20..25 coord MLP 26..31 em MLP 32 eatt_w 33 eatt_b 34 ffn_ln_g 35 ffn_ln_b
  // 36..41 wf MLP 42..47 ffn MLP
  float* outTok  = (float*)d_out;
  float* outGeo  = outTok + (long)NN * DD;
  float* outEdge = outGeo + (long)NN * 3;

  float* w = (float*)d_ws;
  long o = 0;
  float* xln   = w + o; o += (long)NN * DD;
  float* qkv   = w + o; o += (long)NN * 384;
  float* big2  = w + o; o += (long)NKR * DD;     // eln, later em hidden
  float* big1  = w + o; o += (long)NKR * 256;    // qkve, later em_out + ffn hidden
  float* attnW = w + o; o += (long)NN * DD;
  float* scec  = w + o; o += (long)NN * 256;
  float* outh  = w + o; o += (long)NN * DD;
  float* gateb = w + o; o += NN;
  float* c0    = w + o; o += 16;
  float* proj  = w + o; o += (long)NKR * 3;
  float* rnb   = w + o; o += NKR;
  float* ff    = w + o; o += (long)NKR * DD;
  float* Vg    = w + o; o += 40;
  float* c2    = w + o; o += 16;
  float* projg = w + o; o += (long)NN * 3;
  float* gf    = w + o; o += (long)NN * DD;
  float* bt    = w + o; o += (long)NN * DD;
  float* eln    = big2;
  float* em_h   = big2;
  float* qkve   = big1;
  float* em_out = big1;
  float* ffn_h  = big1 + (long)NKR * DD;

  // ---- attention scalar path
  rowln_kernel<<<NN, 128, 0, stream>>>(tok, PR(0), PR(1), xln, 128, 0);
  gemm_kernel<<<dim3(32, 3), 256, 0, stream>>>(xln, 128, PR(2), 384, PR(3), nullptr,
      qkv, 384, NN, 384, 128, 0, nullptr, nullptr, nullptr);
  rowln_kernel<<<NKR, 128, 0, stream>>>(edge, PR(4), PR(5), eln, 128, 0);
  gemm_kernel<<<dim3(512, 2), 256, 0, stream>>>(eln, 128, PR(6), 256, PR(7), nullptr,
      qkve, 256, NKR, 256, 128, 0, nullptr, nullptr, nullptr);
  attn_softmax_kernel<<<NN, 128, 0, stream>>>(qkv, qkve, nidx, nmask, PR(8), PR(9), attnW);
  context_kernel<<<NN, 256, 0, stream>>>(qkv, qkve, nidx, attnW, scec);
  gemm_kernel<<<dim3(32, 1), 256, 0, stream>>>(scec, 256, PR(10), 128, PR(11), nullptr,
      outh, 128, NN, 128, 256, 0, nullptr, nullptr, nullptr);
  rowln_kernel<<<NN, 128, 0, stream>>>(outh, PR(12), PR(13), outh, 128, 1);
  gemm_kernel<<<dim3(32, 1), 256, 0, stream>>>(outh, 128, PR(14), 128, PR(15), tok,
      outTok, 128, NN, 128, 128, 0, nullptr, nullptr, nullptr);

  // ---- geometric branch (collapses exactly to centroid: mean of the 8 sign-frames is 0)
  gate_kernel<<<NN / 8, 256, 0, stream>>>(tok, PR(16), PR(17), gateb);
  center_kernel<<<BB, 256, 0, stream>>>(geo, c0);
  geo_out_kernel<<<NN / 256, 256, 0, stream>>>(geo, gateb, c0, outGeo);

  // ---- edge module
  node_frames_kernel<<<NN / 256, 256, 0, stream>>>(outGeo, nidx, nmask, proj, rnb);
  frame_mlp_kernel<<<NKR / 8, 256, 0, stream>>>(0, proj, rnb,
      PR(20), PR(21), PR(22), PR(23), PR(24), PR(25), ff);
  gemm_kernel<<<dim3(512, 1), 256, 0, stream>>>(nullptr, 0, PR(26), 128, PR(27), nullptr,
      em_h, 128, NKR, 128, 384, 1, outTok, ff, nidx);
  rowln_kernel<<<NKR, 128, 0, stream>>>(em_h, PR(28), PR(29), em_h, 128, 1);
  gemm_kernel<<<dim3(512, 1), 256, 0, stream>>>(em_h, 128, PR(30), 128, PR(31), nullptr,
      em_out, 128, NKR, 128, 128, 0, nullptr, nullptr, nullptr);
  edge_gate_kernel<<<NKR, 128, 0, stream>>>(em_out, edge, PR(32), PR(33), outEdge);

  // ---- frame-averaged FFN
  rowln_kernel<<<NN, 128, 0, stream>>>(outTok, PR(34), PR(35), bt, 128, 0);
  graph_frames_kernel<<<BB, 256, 0, stream>>>(outGeo, Vg, c2);
  projg_kernel<<<NN / 256, 256, 0, stream>>>(outGeo, Vg, c2, projg);
  frame_mlp_kernel<<<NN / 8, 256, 0, stream>>>(1, projg, nullptr,
      PR(36), PR(37), PR(38), PR(39), PR(40), PR(41), gf);
  gemm_kernel<<<dim3(32, 4), 256, 0, stream>>>(nullptr, 0, PR(42), 512, PR(43), nullptr,
      ffn_h, 512, NN, 512, 256, 2, bt, gf, nullptr);
  rowln_kernel<<<NN, 128, 0, stream>>>(ffn_h, PR(44), PR(45), ffn_h, 512, 1);
  gemm_kernel<<<dim3(32, 1), 256, 0, stream>>>(ffn_h, 512, PR(46), 128, PR(47), outTok,
      outTok, 128, NN, 128, 512, 0, nullptr, nullptr, nullptr);
#undef PR
}
